// GraphClassificationModel_79714593014205
// MI455X (gfx1250) — compile-verified, matrix operand order checked by reading, not run on hardware
//
#include <hip/hip_runtime.h>
#include <hip/hip_bf16.h>

typedef __attribute__((ext_vector_type(16))) __bf16 v16bf;
typedef __attribute__((ext_vector_type(8)))  __bf16 v8bf;
typedef __attribute__((ext_vector_type(8)))  float  v8f;
typedef __attribute__((ext_vector_type(4)))  float  v4f;

#define HID 128
#define NGRAPHS 512
#define OUTC 10

// ---------------------------------------------------------------------------
// Degree accumulation: one thread per edge, deg[dst] += 1
// ---------------------------------------------------------------------------
__global__ __launch_bounds__(256) void k_degree(const long long* __restrict__ dst,
                                                float* __restrict__ deg, int nedges) {
    int e = blockIdx.x * 256 + threadIdx.x;
    if (e >= nedges) return;
    atomicAdd(&deg[(int)dst[e]], 1.0f);
}

// dinv = rsqrt(deg + 1)   (in-place over deg buffer)
__global__ __launch_bounds__(256) void k_dinv(float* __restrict__ deg, int n) {
    int i = blockIdx.x * 256 + threadIdx.x;
    if (i >= n) return;
    deg[i] = rsqrtf(deg[i] + 1.0f);
}

// ---------------------------------------------------------------------------
// GEMM: H[n,128] = X[n,128] @ W[128,128] via v_wmma_f32_16x16x32_bf16.
// 8 waves/block; wave w computes rows [blk*128 + w*16, +16) x all 128 cols.
// W is staged transposed (n-major) in LDS as bf16 for b128 DS fragment loads.
// ---------------------------------------------------------------------------
__global__ __launch_bounds__(256) void k_gemm_xw(const float* __restrict__ X,
                                                 const float* __restrict__ W,
                                                 float* __restrict__ H, int nrows) {
    __shared__ __align__(16) __bf16 Wt[HID * HID];   // 32 KB, layout [n][k]

    int tid = threadIdx.x;
    // Stage W (row-major [k][n] fp32) -> LDS transposed bf16
    for (int i = tid; i < HID * HID; i += 256) {
        int k = i >> 7, n = i & 127;
        Wt[n * HID + k] = (__bf16)W[i];
    }
    __syncthreads();

    int wave = tid >> 5, lane = tid & 31;
    int lrow = lane & 15, lhalf = lane >> 4;
    int rowBase = blockIdx.x * 128 + wave * 16;

    int arow = rowBase + lrow;
    if (arow >= nrows) arow = nrows - 1;             // clamp loads; stores guarded
    const float* xrow = X + (long)arow * HID;

    v8f acc[8];
#pragma unroll
    for (int t = 0; t < 8; ++t) acc[t] = (v8f){0.f,0.f,0.f,0.f,0.f,0.f,0.f,0.f};

#pragma unroll
    for (int kk = 0; kk < 4; ++kk) {
        int kb = kk * 32 + lhalf * 8;
        // A fragment: lane holds row (lane&15), K = kb..kb+7 and kb+16..kb+23
        v4f a0 = *(const v4f*)(xrow + kb);
        v4f a1 = *(const v4f*)(xrow + kb + 4);
        v4f a2 = *(const v4f*)(xrow + kb + 16);
        v4f a3 = *(const v4f*)(xrow + kb + 20);
        v16bf afrag;
        afrag[0]=(__bf16)a0.x; afrag[1]=(__bf16)a0.y; afrag[2]=(__bf16)a0.z; afrag[3]=(__bf16)a0.w;
        afrag[4]=(__bf16)a1.x; afrag[5]=(__bf16)a1.y; afrag[6]=(__bf16)a1.z; afrag[7]=(__bf16)a1.w;
        afrag[8]=(__bf16)a2.x; afrag[9]=(__bf16)a2.y; afrag[10]=(__bf16)a2.z; afrag[11]=(__bf16)a2.w;
        afrag[12]=(__bf16)a3.x; afrag[13]=(__bf16)a3.y; afrag[14]=(__bf16)a3.z; afrag[15]=(__bf16)a3.w;

#pragma unroll
        for (int t = 0; t < 8; ++t) {
            int n = t * 16 + lrow;                   // lane holds column n
            const __bf16* wp = &Wt[n * HID + kb];
            v8bf blo = *(const v8bf*)(wp);
            v8bf bhi = *(const v8bf*)(wp + 16);
            v16bf bfrag;
#pragma unroll
            for (int e2 = 0; e2 < 8; ++e2) { bfrag[e2] = blo[e2]; bfrag[8 + e2] = bhi[e2]; }
            acc[t] = __builtin_amdgcn_wmma_f32_16x16x32_bf16(
                false, afrag, false, bfrag, (short)0, acc[t], false, false);
        }
    }

    // C/D layout: lanes 0-15 -> N=lane, M = vgpr; lanes 16-31 -> N=lane-16, M = 8+vgpr
#pragma unroll
    for (int t = 0; t < 8; ++t) {
        int n = t * 16 + lrow;
#pragma unroll
        for (int v = 0; v < 8; ++v) {
            int r = rowBase + lhalf * 8 + v;
            if (r < nrows) H[(long)r * HID + n] = acc[t][v];
        }
    }
}

// ---------------------------------------------------------------------------
// Edge scatter: one wave per edge, 4 channels per lane.
// agg[dst,:] += h[src,:] * dinv[src]*dinv[dst]
// h (51.2 MB) is L2-resident, so gathers + atomics stay on-die.
// ---------------------------------------------------------------------------
__global__ __launch_bounds__(256) void k_scatter(const float* __restrict__ H,
                                                 const long long* __restrict__ src,
                                                 const long long* __restrict__ dst,
                                                 const float* __restrict__ dinv,
                                                 float* __restrict__ AGG, int nedges) {
    int gid = blockIdx.x * 256 + threadIdx.x;
    int e = gid >> 5;
    if (e >= nedges) return;
    int lane = threadIdx.x & 31;
    int s = (int)src[e], d = (int)dst[e];
    float coef = dinv[s] * dinv[d];
    v4f hv = *(const v4f*)(H + (long)s * HID + lane * 4);
    float* ap = AGG + (long)d * HID + lane * 4;
    atomicAdd(ap + 0, hv.x * coef);
    atomicAdd(ap + 1, hv.y * coef);
    atomicAdd(ap + 2, hv.z * coef);
    atomicAdd(ap + 3, hv.w * coef);
}

// ---------------------------------------------------------------------------
// Epilogue: out = agg + h*dinv^2 + b, optional ReLU (in-place over agg)
// ---------------------------------------------------------------------------
__global__ __launch_bounds__(256) void k_epilogue(float* __restrict__ AGG,
                                                  const float* __restrict__ H,
                                                  const float* __restrict__ dinv,
                                                  const float* __restrict__ bias,
                                                  int n, int relu) {
    long idx = (long)blockIdx.x * 256 + threadIdx.x;
    if (idx >= (long)n * HID) return;
    int i = (int)(idx >> 7), c = (int)(idx & 127);
    float di = dinv[i];
    float v = AGG[idx] + H[idx] * di * di + bias[c];
    AGG[idx] = relu ? fmaxf(v, 0.0f) : v;
}

// ---------------------------------------------------------------------------
// Pooling
// ---------------------------------------------------------------------------
__global__ __launch_bounds__(256) void k_pool_sum(const float* __restrict__ ACT,
                                                  const long long* __restrict__ batch,
                                                  float* __restrict__ pooled, int n) {
    long idx = (long)blockIdx.x * 256 + threadIdx.x;
    if (idx >= (long)n * HID) return;
    int i = (int)(idx >> 7), c = (int)(idx & 127);
    int g = (int)batch[i];
    atomicAdd(&pooled[g * HID + c], ACT[idx]);
}

__global__ __launch_bounds__(256) void k_pool_cnt(const long long* __restrict__ batch,
                                                  float* __restrict__ counts, int n) {
    int i = blockIdx.x * 256 + threadIdx.x;
    if (i >= n) return;
    atomicAdd(&counts[(int)batch[i]], 1.0f);
}

// ---------------------------------------------------------------------------
// Tiny MLP head: one block (128 threads) per graph.
// out = relu(mean @ W1 + b1) @ W2 + b2
// ---------------------------------------------------------------------------
__global__ __launch_bounds__(128) void k_mlp(const float* __restrict__ pooled,
                                             const float* __restrict__ counts,
                                             const float* __restrict__ W1,
                                             const float* __restrict__ b1,
                                             const float* __restrict__ W2,
                                             const float* __restrict__ b2,
                                             float* __restrict__ out) {
    __shared__ float p[HID];
    __shared__ float hid[HID];
    int g = blockIdx.x, t = threadIdx.x;
    float cnt = fmaxf(counts[g], 1.0f);
    p[t] = pooled[g * HID + t] / cnt;
    __syncthreads();
    float acc = b1[t];
#pragma unroll 4
    for (int k = 0; k < HID; ++k) acc += p[k] * W1[k * HID + t];
    hid[t] = fmaxf(acc, 0.0f);
    __syncthreads();
    if (t < OUTC) {
        float o = b2[t];
#pragma unroll 4
        for (int k = 0; k < HID; ++k) o += hid[k] * W2[k * OUTC + t];
        out[g * OUTC + t] = o;
    }
}

// ---------------------------------------------------------------------------
// Launch
// ---------------------------------------------------------------------------
extern "C" void kernel_launch(void* const* d_in, const int* in_sizes, int n_in,
                              void* d_out, int out_size, void* d_ws, size_t ws_size,
                              hipStream_t stream) {
    const float*     x    = (const float*)d_in[0];
    const long long* ei   = (const long long*)d_in[1];
    const long long* batch= (const long long*)d_in[2];
    const float* Wc0 = (const float*)d_in[3];  const float* bc0 = (const float*)d_in[4];
    const float* Wc1 = (const float*)d_in[5];  const float* bc1 = (const float*)d_in[6];
    const float* Wc2 = (const float*)d_in[7];  const float* bc2 = (const float*)d_in[8];
    const float* W1  = (const float*)d_in[9];  const float* b1  = (const float*)d_in[10];
    const float* W2  = (const float*)d_in[11]; const float* b2  = (const float*)d_in[12];
    float* out = (float*)d_out;

    const int n      = in_sizes[0] / HID;      // 100000
    const int nedges = in_sizes[1] / 2;        // 1600000
    const long long* src = ei;
    const long long* dst = ei + nedges;

    // Workspace carve-up
    char* ws = (char*)d_ws;
    size_t off = 0;
    auto take = [&](size_t bytes) { void* p = ws + off; off += (bytes + 255) & ~(size_t)255; return p; };
    float* dinv   = (float*)take((size_t)n * 4);
    float* bufH   = (float*)take((size_t)n * HID * 4);
    float* bufA   = (float*)take((size_t)n * HID * 4);
    float* bufB   = (float*)take((size_t)n * HID * 4);
    float* pooled = (float*)take((size_t)NGRAPHS * HID * 4);
    float* counts = (float*)take((size_t)NGRAPHS * 4);

    const int nc_blocks   = (n * HID + 255) / 256;       // node*channel grids
    const int node_blocks = (n + 255) / 256;
    const int edge_blocks = (nedges + 255) / 256;
    const int wave_blocks = (nedges * 32 + 255) / 256;   // wave-per-edge scatter
    const int gemm_blocks = (n + 127) / 128;

    // Degrees -> dinv
    hipMemsetAsync(dinv, 0, (size_t)n * 4, stream);
    k_degree<<<edge_blocks, 256, 0, stream>>>(dst, dinv, nedges);
    k_dinv<<<node_blocks, 256, 0, stream>>>(dinv, n);

    // Layer 1: x -> bufA
    k_gemm_xw<<<gemm_blocks, 256, 0, stream>>>(x, Wc0, bufH, n);
    hipMemsetAsync(bufA, 0, (size_t)n * HID * 4, stream);
    k_scatter<<<wave_blocks, 256, 0, stream>>>(bufH, src, dst, dinv, bufA, nedges);
    k_epilogue<<<nc_blocks, 256, 0, stream>>>(bufA, bufH, dinv, bc0, n, 1);

    // Layer 2: bufA -> bufB
    k_gemm_xw<<<gemm_blocks, 256, 0, stream>>>(bufA, Wc1, bufH, n);
    hipMemsetAsync(bufB, 0, (size_t)n * HID * 4, stream);
    k_scatter<<<wave_blocks, 256, 0, stream>>>(bufH, src, dst, dinv, bufB, nedges);
    k_epilogue<<<nc_blocks, 256, 0, stream>>>(bufB, bufH, dinv, bc1, n, 1);

    // Layer 3: bufB -> bufA (no ReLU)
    k_gemm_xw<<<gemm_blocks, 256, 0, stream>>>(bufB, Wc2, bufH, n);
    hipMemsetAsync(bufA, 0, (size_t)n * HID * 4, stream);
    k_scatter<<<wave_blocks, 256, 0, stream>>>(bufH, src, dst, dinv, bufA, nedges);
    k_epilogue<<<nc_blocks, 256, 0, stream>>>(bufA, bufH, dinv, bc2, n, 0);

    // Pool + head
    hipMemsetAsync(pooled, 0, (size_t)NGRAPHS * HID * 4, stream);
    hipMemsetAsync(counts, 0, (size_t)NGRAPHS * 4, stream);
    k_pool_sum<<<nc_blocks, 256, 0, stream>>>(bufA, batch, pooled, n);
    k_pool_cnt<<<node_blocks, 256, 0, stream>>>(batch, counts, n);
    k_mlp<<<NGRAPHS, 128, 0, stream>>>(pooled, counts, W1, b1, W2, b2, out);
}
